// MultiHeadAttention_54631984005149
// MI455X (gfx1250) — matrix operand order repaired and checked by reading, not compiled, not run on hardware
//
#include <hip/hip_runtime.h>

// Shapes from the reference
#define Bsz 2
#define Ssz 2048
#define Dsz 1024
#define Hsz 16
#define DH  64
#define Msz (Bsz * Ssz)   // 4096 rows for projection GEMMs

typedef __attribute__((ext_vector_type(16))) _Float16 v16h;
typedef __attribute__((ext_vector_type(8)))  _Float16 v8h;
typedef __attribute__((ext_vector_type(4)))  _Float16 v4h;
typedef __attribute__((ext_vector_type(8)))  float    v8f;
typedef __attribute__((ext_vector_type(4)))  int      v4i;

// ---------------------------------------------------------------------------
// CDNA5 async global->LDS copy (GLOBAL_LOAD_ASYNC_TO_LDS_B128, ASYNCcnt).
// Probe result: builtin exists and takes (AS1 v4i*, AS3 v4i*, imm, imm)
// ("__device__" in the clang diagnostic == address_space(1) pointee).
// Guarded by __has_builtin so the file still compiles (register fallback)
// on toolchains without the builtin.
// ---------------------------------------------------------------------------
#if defined(__HIP_DEVICE_COMPILE__) && __has_builtin(__builtin_amdgcn_global_load_async_to_lds_b128)
#define HAVE_ASYNC_LDS 1
typedef __attribute__((address_space(1))) v4i g_v4i;
typedef __attribute__((address_space(3))) v4i l_v4i;
#endif

static __device__ __forceinline__ void cp_async16(const _Float16* gsrc, _Float16* ldst) {
#ifdef HAVE_ASYNC_LDS
    __builtin_amdgcn_global_load_async_to_lds_b128(
        (g_v4i*)(void*)gsrc, (l_v4i*)(void*)ldst, 0, 0);
#else
    *(v8h*)ldst = *(const v8h*)gsrc;
#endif
}

static __device__ __forceinline__ void async_wait_all() {
#ifdef HAVE_ASYNC_LDS
#if __has_builtin(__builtin_amdgcn_s_wait_asynccnt)
    __builtin_amdgcn_s_wait_asynccnt(0);
#else
    asm volatile("s_wait_asynccnt 0x0" ::: "memory");
#endif
#endif
}

// ---------------------------------------------------------------------------
// Fragment helpers.
// CDNA5 WMMA f16 operand layouts (CDNA5 ISA):
//  A (16x32, MxK): lane L holds row m=L&15; halves K in [k0,k0+8) and
//                  [k0+16,k0+24) where k0 = 8*(L>>4)  -> two 16B chunks.
//  B (32x16, KxN): lane L holds col n=L&15; halves K in [k0,k0+8) and
//                  [k0+16,k0+24) where k0 = 8*(L>>4)  -> two 16B chunks.
//  C/D (16x16 f32): lane L col n=L&15; VGPR j holds row m = j + 8*(L>>4).
// ---------------------------------------------------------------------------
static __device__ __forceinline__ v16h ldfrag(const _Float16* p0, const _Float16* p1) {
    v8h lo = *(const v8h*)p0;
    v8h hi = *(const v8h*)p1;
    return __builtin_shufflevector(lo, hi, 0,1,2,3,4,5,6,7,8,9,10,11,12,13,14,15);
}

static __device__ __forceinline__ v8f wmma_f16(v16h a, v16h b, v8f c) {
    return __builtin_amdgcn_wmma_f32_16x16x32_f16(false, a, false, b, (short)0, c, false, false);
}

// ---------------------------------------------------------------------------
// f32 -> f16 pre-conversion (x and the 4 weight matrices, done once so the
// GEMM hot loops are pure f16 byte-copies + WMMA). n must be multiple of 1024.
// ---------------------------------------------------------------------------
__global__ __launch_bounds__(256) void cvt_kernel(const float* __restrict__ src,
                                                  _Float16* __restrict__ dst) {
    const int i = (blockIdx.x * 256 + threadIdx.x) * 4;
    float4 f = *(const float4*)(src + i);
    v4h h; h.x = (_Float16)f.x; h.y = (_Float16)f.y;
    h.z = (_Float16)f.z; h.w = (_Float16)f.w;
    *(v4h*)(dst + i) = h;
}

// ---------------------------------------------------------------------------
// All-f16 GEMM: C[M x 1024] = A[M x 1024] * Wh[1024 x 1024]^T (+bias[+resid])
// MODE 0: output f16 written in [B,H,S,DH] head layout (Q/K/V).
// MODE 1: output f32 y = C + bias + residual(x), row-major.
// Block 128 threads (4 waves). WG tile 128x64, wave tile 32x64 (8 WMMA per
// 32-K step), tiles staged to LDS with async b128 copies.
// ---------------------------------------------------------------------------
#define LDA 40   // LDS row stride in halfs (80B, 16B-aligned rows)

template <int MODE>
__global__ __launch_bounds__(128) void gemm_f16(const _Float16* __restrict__ A,
                                                const _Float16* __restrict__ Wh,
                                                const float* __restrict__ bias,
                                                const float* __restrict__ resid,
                                                void* __restrict__ Cout) {
    __shared__ _Float16 As[128 * LDA];
    __shared__ _Float16 Bs[64 * LDA];

    const int tid  = threadIdx.x;
    const int wave = tid >> 5;
    const int lane = tid & 31;
    const int lane_lo = lane & 15;
    const int lane_hi = lane >> 4;

    const int block_m = blockIdx.y * 128;
    const int block_n = blockIdx.x * 64;

    v8f acc[2][4] = {};

    for (int kt = 0; kt < Dsz / 32; ++kt) {
        const int k0 = kt * 32;

        // ---- stage A tile (128 rows x 32 halfs): thread t copies row t ----
        {
            const _Float16* src = A + (size_t)(block_m + tid) * Dsz + k0;
            _Float16* dst = &As[tid * LDA];
            #pragma unroll
            for (int u = 0; u < 4; ++u) cp_async16(src + u * 8, dst + u * 8);
        }
        // ---- stage W tile as Bs[n][k] (64 rows x 32 halfs) ----
        {
            const int row = tid >> 1, off = (tid & 1) * 16;
            const _Float16* src = Wh + (size_t)(block_n + row) * Dsz + k0 + off;
            _Float16* dst = &Bs[row * LDA + off];
            cp_async16(src, dst);
            cp_async16(src + 8, dst + 8);
        }
        async_wait_all();
        __syncthreads();

        v16h afrag[2], bfrag[4];
        #pragma unroll
        for (int i = 0; i < 2; ++i) {
            const int m  = wave * 32 + i * 16 + lane_lo;
            const int kk = 8 * lane_hi;
            afrag[i] = ldfrag(&As[m * LDA + kk], &As[m * LDA + kk + 16]);
        }
        #pragma unroll
        for (int j = 0; j < 4; ++j) {
            const int n  = j * 16 + lane_lo;
            const int kk = 8 * lane_hi;
            bfrag[j] = ldfrag(&Bs[n * LDA + kk], &Bs[n * LDA + kk + 16]);
        }
        #pragma unroll
        for (int i = 0; i < 2; ++i)
            #pragma unroll
            for (int j = 0; j < 4; ++j)
                acc[i][j] = wmma_f16(afrag[i], bfrag[j], acc[i][j]);
        __syncthreads();
    }

    // ---- epilogue ----
    #pragma unroll
    for (int i = 0; i < 2; ++i) {
        #pragma unroll
        for (int j = 0; j < 4; ++j) {
            const int gn = block_n + j * 16 + lane_lo;
            const float bv = bias[gn];
            #pragma unroll
            for (int e = 0; e < 8; ++e) {
                const int gm = block_m + wave * 32 + i * 16 + e + 8 * lane_hi;
                float v = acc[i][j][e] + bv;
                if (MODE == 0) {
                    const int b = gm >> 11, s = gm & (Ssz - 1);
                    const int h = gn >> 6,  d = gn & (DH - 1);
                    ((_Float16*)Cout)[((size_t)((b * Hsz + h) * Ssz + s)) * DH + d] = (_Float16)v;
                } else {
                    v += resid[(size_t)gm * Dsz + gn];
                    ((float*)Cout)[(size_t)gm * Dsz + gn] = v;
                }
            }
        }
    }
}

// ---------------------------------------------------------------------------
// Flash attention, causal. Grid: (S/64, B*H). Block: 128 threads (4 waves).
// Wave w handles 16 query rows; 64-key chunks staged to LDS (K via async
// b128 copies, V transposed through registers). scores = (Q K^T)*0.125,
// online softmax in C-layout, P via per-wave LDS, ctx += P V.
// ---------------------------------------------------------------------------
#define LDK 72   // LDS row stride in halfs (144B, 16B-aligned)

__global__ __launch_bounds__(128) void attn_kernel(const _Float16* __restrict__ Q,
                                                   const _Float16* __restrict__ Kd,
                                                   const _Float16* __restrict__ Vd,
                                                   _Float16* __restrict__ ctx) {
    __shared__ _Float16 Ks[64 * LDK];        // [key][dh]
    __shared__ _Float16 Vt[64 * LDK];        // [dh][key]
    __shared__ _Float16 Pw[4][16 * LDK];     // per-wave P scratch [row][key]

    const int tid  = threadIdx.x;
    const int wave = tid >> 5;
    const int lane = tid & 31;
    const int lane_lo = lane & 15;
    const int lane_hi = lane >> 4;

    const int bh = blockIdx.y;               // b*H + h
    const int qb = blockIdx.x;               // 64-row query block
    const size_t base = (size_t)bh * Ssz * DH;
    const int qrow0 = qb * 64 + wave * 16;

    // Q fragments held in registers for the whole key loop (DH=64 -> 2 frags)
    v16h qa[2];
    #pragma unroll
    for (int i = 0; i < 2; ++i) {
        const int k0 = 8 * lane_hi + i * 32;
        const _Float16* p = Q + base + (size_t)(qrow0 + lane_lo) * DH + k0;
        qa[i] = ldfrag(p, p + 16);
    }

    v8f ctxa[4] = {};
    v8f mrow, lrow;
    #pragma unroll
    for (int e = 0; e < 8; ++e) { mrow[e] = -1.0e30f; lrow[e] = 0.0f; }

    const float LOG2E = 1.4426950408889634f;

    for (int c = 0; c <= qb; ++c) {
        const int kbase = c * 64;
        // ---- stage Ks row-major via async copy (pure f16 byte copy) ----
        {
            const int row = tid >> 1, off = (tid & 1) * 32;
            const _Float16* src = Kd + base + (size_t)(kbase + row) * DH + off;
            _Float16* dst = &Ks[row * LDK + off];
            #pragma unroll
            for (int u = 0; u < 4; ++u) cp_async16(src + u * 8, dst + u * 8);
        }
        // ---- stage Vt transposed through registers ----
        {
            const int key   = tid >> 1;
            const int dpart = (tid & 1) * 32;
            const _Float16* src = Vd + base + (size_t)(kbase + key) * DH + dpart;
            #pragma unroll
            for (int u = 0; u < 4; ++u) {
                v8h v = *(const v8h*)(src + 8 * u);
                #pragma unroll
                for (int e = 0; e < 8; ++e)
                    Vt[(dpart + 8 * u + e) * LDK + key] = v[e];
            }
        }
        async_wait_all();
        __syncthreads();

        // ---- scores: 4 key-subtiles x 2 K-steps of WMMA ----
        v8f s[4];
        #pragma unroll
        for (int ns = 0; ns < 4; ++ns) {
            const int kcol = ns * 16 + lane_lo;
            v8f a = {};
            #pragma unroll
            for (int ks = 0; ks < 2; ++ks) {
                const int kk = ks * 32 + 8 * lane_hi;
                v16h b = ldfrag(&Ks[kcol * LDK + kk], &Ks[kcol * LDK + kk + 16]);
                a = wmma_f16(qa[ks], b, a);
            }
            // scale + causal mask in C-layout
            const int key_g = kbase + ns * 16 + lane_lo;
            #pragma unroll
            for (int e = 0; e < 8; ++e) {
                const int row_g = qrow0 + e + 8 * lane_hi;
                float sv = a[e] * 0.125f;              // 1/sqrt(DH)
                if (key_g > row_g) sv = -1.0e30f;
                a[e] = sv;
            }
            s[ns] = a;
        }

        // ---- online softmax: row reductions across 16 lanes ----
        v8f rm;
        #pragma unroll
        for (int e = 0; e < 8; ++e) {
            float v = s[0][e];
            v = fmaxf(v, s[1][e]); v = fmaxf(v, s[2][e]); v = fmaxf(v, s[3][e]);
            v = fmaxf(v, __shfl_xor(v, 1, 32));
            v = fmaxf(v, __shfl_xor(v, 2, 32));
            v = fmaxf(v, __shfl_xor(v, 4, 32));
            v = fmaxf(v, __shfl_xor(v, 8, 32));
            rm[e] = v;
        }
        v8f mnew, alpha;
        #pragma unroll
        for (int e = 0; e < 8; ++e) {
            mnew[e]  = fmaxf(mrow[e], rm[e]);
            alpha[e] = exp2f((mrow[e] - mnew[e]) * LOG2E);
        }
        v8f rs = {};
        #pragma unroll
        for (int ns = 0; ns < 4; ++ns)
            #pragma unroll
            for (int e = 0; e < 8; ++e) {
                float p = exp2f((s[ns][e] - mnew[e]) * LOG2E);
                s[ns][e] = p;
                rs[e] += p;
            }
        #pragma unroll
        for (int e = 0; e < 8; ++e) {
            float v = rs[e];
            v += __shfl_xor(v, 1, 32);
            v += __shfl_xor(v, 2, 32);
            v += __shfl_xor(v, 4, 32);
            v += __shfl_xor(v, 8, 32);
            lrow[e] = lrow[e] * alpha[e] + v;
            mrow[e] = mnew[e];
        }
        #pragma unroll
        for (int ns = 0; ns < 4; ++ns)
            #pragma unroll
            for (int e = 0; e < 8; ++e)
                ctxa[ns][e] *= alpha[e];

        // ---- spill P to per-wave LDS (C-layout -> row-major) ----
        _Float16* pw = &Pw[wave][0];
        #pragma unroll
        for (int ns = 0; ns < 4; ++ns) {
            const int col = ns * 16 + lane_lo;
            #pragma unroll
            for (int e = 0; e < 8; ++e)
                pw[(e + 8 * lane_hi) * LDK + col] = (_Float16)s[ns][e];
        }
        // same-wave DS ops are in order; compiler inserts s_wait_dscnt

        // ---- ctx += P x V : 4 dh-subtiles x 2 K-steps ----
        #pragma unroll
        for (int dsub = 0; dsub < 4; ++dsub) {
            #pragma unroll
            for (int ks = 0; ks < 2; ++ks) {
                const int k0 = ks * 32 + 8 * lane_hi;
                v16h pa = ldfrag(&pw[lane_lo * LDK + k0], &pw[lane_lo * LDK + k0 + 16]);
                const int d  = dsub * 16 + lane_lo;
                const int kb = ks * 32 + 8 * lane_hi;
                v16h vb = ldfrag(&Vt[d * LDK + kb], &Vt[d * LDK + kb + 16]);
                ctxa[dsub] = wmma_f16(pa, vb, ctxa[dsub]);
            }
        }
        __syncthreads();
    }

    // ---- epilogue: normalize by l, store f16 ctx in [B,S,D] row-major ----
    const int b = bh >> 4, h = bh & (Hsz - 1);
    #pragma unroll
    for (int ns = 0; ns < 4; ++ns) {
        const int col = h * DH + ns * 16 + lane_lo;
        #pragma unroll
        for (int e = 0; e < 8; ++e) {
            const int srow = qrow0 + e + 8 * lane_hi;
            float v = ctxa[ns][e] / lrow[e];
            ctx[((size_t)(b * Ssz + srow)) * Dsz + col] = (_Float16)v;
        }
    }
}

// ---------------------------------------------------------------------------
// LayerNorm: one block (256 threads) per row of y[4096][1024].
// ---------------------------------------------------------------------------
__global__ __launch_bounds__(256) void ln_kernel(const float* __restrict__ y,
                                                 const float* __restrict__ g,
                                                 const float* __restrict__ bta,
                                                 float* __restrict__ out) {
    __shared__ float s_sum[8], s_sq[8];
    const int row = blockIdx.x;
    const int tid = threadIdx.x;
    const float* yr = y + (size_t)row * Dsz;

    float sum = 0.f, sq = 0.f;
    #pragma unroll
    for (int i = 0; i < 4; ++i) {
        float v = yr[tid + i * 256];
        sum += v; sq += v * v;
    }
    #pragma unroll
    for (int m = 1; m < 32; m <<= 1) {
        sum += __shfl_xor(sum, m, 32);
        sq  += __shfl_xor(sq,  m, 32);
    }
    if ((tid & 31) == 0) { s_sum[tid >> 5] = sum; s_sq[tid >> 5] = sq; }
    __syncthreads();
    float tsum = 0.f, tsq = 0.f;
    #pragma unroll
    for (int w = 0; w < 8; ++w) { tsum += s_sum[w]; tsq += s_sq[w]; }
    const float mean = tsum * (1.0f / Dsz);
    const float var  = tsq * (1.0f / Dsz) - mean * mean;
    const float rstd = rsqrtf(var + 1e-5f);
    #pragma unroll
    for (int i = 0; i < 4; ++i) {
        const int c = tid + i * 256;
        out[(size_t)row * Dsz + c] = (yr[c] - mean) * rstd * g[c] + bta[c];
    }
}

// ---------------------------------------------------------------------------
// Launch. Workspace layout (bytes, requires ws_size >= 64MB):
//   xh  f16 [4096x1024]  :  0 ..  8MB
//   Wqh/Wkh/Wvh/Woh f16  :  8 .. 16MB (2MB each)
//   Q/K/V f16 [B,H,S,DH] : 16 .. 40MB (8MB each)
//   ctx f16 [B,S,D]      : 40 .. 48MB
//   y   f32 [B,S,D]      : 48 .. 64MB
// ---------------------------------------------------------------------------
extern "C" void kernel_launch(void* const* d_in, const int* in_sizes, int n_in,
                              void* d_out, int out_size, void* d_ws, size_t ws_size,
                              hipStream_t stream) {
    const float* x    = (const float*)d_in[0];
    const float* Wq   = (const float*)d_in[1];
    const float* bq   = (const float*)d_in[2];
    const float* Wk   = (const float*)d_in[3];
    const float* bk   = (const float*)d_in[4];
    const float* Wv   = (const float*)d_in[5];
    const float* bv   = (const float*)d_in[6];
    const float* Wo   = (const float*)d_in[7];
    const float* bo   = (const float*)d_in[8];
    const float* ln_g = (const float*)d_in[9];
    const float* ln_b = (const float*)d_in[10];

    char* ws = (char*)d_ws;
    const size_t MB = 1024 * 1024;
    _Float16* xh    = (_Float16*)(ws);
    _Float16* wqh   = (_Float16*)(ws + 8 * MB);
    _Float16* wkh   = (_Float16*)(ws + 10 * MB);
    _Float16* wvh   = (_Float16*)(ws + 12 * MB);
    _Float16* woh   = (_Float16*)(ws + 14 * MB);
    _Float16* qws   = (_Float16*)(ws + 16 * MB);
    _Float16* kws   = (_Float16*)(ws + 24 * MB);
    _Float16* vws   = (_Float16*)(ws + 32 * MB);
    _Float16* ctxws = (_Float16*)(ws + 40 * MB);
    float*    yws   = (float*)   (ws + 48 * MB);

    // one-time f32 -> f16 conversions
    cvt_kernel<<<Msz * Dsz / 1024, 256, 0, stream>>>(x,  xh);
    cvt_kernel<<<Dsz * Dsz / 1024, 256, 0, stream>>>(Wq, wqh);
    cvt_kernel<<<Dsz * Dsz / 1024, 256, 0, stream>>>(Wk, wkh);
    cvt_kernel<<<Dsz * Dsz / 1024, 256, 0, stream>>>(Wv, wvh);
    cvt_kernel<<<Dsz * Dsz / 1024, 256, 0, stream>>>(Wo, woh);

    dim3 ggrid(Dsz / 64, Msz / 128);  // (16, 32)
    gemm_f16<0><<<ggrid, 128, 0, stream>>>(xh, wqh, bq, nullptr, qws);
    gemm_f16<0><<<ggrid, 128, 0, stream>>>(xh, wkh, bk, nullptr, kws);
    gemm_f16<0><<<ggrid, 128, 0, stream>>>(xh, wvh, bv, nullptr, vws);

    dim3 agrid(Ssz / 64, Bsz * Hsz);  // (32, 32)
    attn_kernel<<<agrid, 128, 0, stream>>>(qws, kws, vws, ctxws);

    gemm_f16<1><<<ggrid, 128, 0, stream>>>(ctxws, woh, bo, x, yws);

    ln_kernel<<<Msz, 256, 0, stream>>>(yws, ln_g, ln_b, (float*)d_out);
}
